// CARDformer_15547781612088
// MI455X (gfx1250) — matrix operand order repaired and checked by reading, not compiled, not run on hardware
//
#include <hip/hip_runtime.h>
#include <hip/hip_bf16.h>

// ---------------------------------------------------------------------------
// CARDformer forward for MI455X (gfx1250, wave32, WMMA 16x16x32 f16,
// async global->LDS staging).
// ---------------------------------------------------------------------------

typedef __attribute__((ext_vector_type(16))) _Float16 v16h;
typedef __attribute__((ext_vector_type(8)))  _Float16 v8h;
typedef __attribute__((ext_vector_type(8)))  float    v8f;

#define B_      16
#define NVARS_  21
#define H_      64
#define DMODEL_ 512
#define NHEADS_ 8
#define DFF_    2048
#define HEADDIM_ 64
#define MTOT_   (B_ * NVARS_ * H_)      // 21504 rows
#define NGROUP_ (B_ * NVARS_ * NHEADS_) // 2688 attention groups

// ---------------------------------------------------------------------------
// CDNA5 async global->LDS copy (CDNA5 ISA).
// VDST VGPR holds the LDS byte address (low 32 bits of the generic pointer:
// the LDS aperture maps the LDS address to those low bits).
// ---------------------------------------------------------------------------
__device__ __forceinline__ void async_copy_b128(void* lds_dst, const void* gsrc) {
  unsigned l = (unsigned)(size_t)lds_dst;
  unsigned long long g = (unsigned long long)(size_t)gsrc;
  asm volatile("global_load_async_to_lds_b128 %0, %1, off"
               :: "v"(l), "v"(g) : "memory");
}
__device__ __forceinline__ void async_wait_all() {
  asm volatile("s_wait_asynccnt 0x0" ::: "memory");
}

// ---------------------------------------------------------------------------
// WMMA fragment loaders (gfx1250 wave32 layouts, CDNA5 ISA 7.12.2)
// ---------------------------------------------------------------------------
// A: 16x32 f16 row-major tile.  lane m = lane&15; low lanes K 0-15,
// high lanes K 16-31.  Two contiguous 16B LDS reads.
__device__ __forceinline__ v16h load_frag_a(const _Float16* s, int ld,
                                            int row0, int k0, int lane) {
  int m  = row0 + (lane & 15);
  int kb = k0 + ((lane & 16) ? 16 : 0);
  union { v16h v; v8h h[2]; } u;
  u.h[0] = *(const v8h*)(s + m * ld + kb);
  u.h[1] = *(const v8h*)(s + m * ld + kb + 8);
  return u.v;
}

// B: 32x16 f16, tile stored N-major in LDS (s[n*ld + k]).  lanes 0-15 hold
// K 0-15, lanes 16-31 hold K 16-31, ascending within the v16h.
__device__ __forceinline__ v16h load_frag_b(const _Float16* s, int ld,
                                            int n0, int k0, int lane) {
  int n  = n0 + (lane & 15);
  int kb = k0 + ((lane & 16) ? 16 : 0);
  union { v16h v; v8h h[2]; } u;
  u.h[0] = *(const v8h*)(s + n * ld + kb);
  u.h[1] = *(const v8h*)(s + n * ld + kb + 8);
  return u.v;
}

__device__ __forceinline__ v8f wmma_f16(v16h a, v16h b, v8f c) {
  return __builtin_amdgcn_wmma_f32_16x16x32_f16(false, a, false, b,
                                                (short)0, c, false, false);
}

// C/D layout: element r -> row r (lanes 0-15) / row r+8 (lanes 16-31),
// col = lane&15.
__device__ __forceinline__ void store_c_lds(float* s, int ld, int row0,
                                            int n0, int lane, v8f acc,
                                            float scale) {
  int col = n0 + (lane & 15);
  int rb  = row0 + ((lane & 16) ? 8 : 0);
#pragma unroll
  for (int r = 0; r < 8; ++r) s[(rb + r) * ld + col] = acc[r] * scale;
}

// ---------------------------------------------------------------------------
// Generic tiled GEMM: C[M,N] = act(A[M,K] @ W[K,N] + bias)   (f32 in/out,
// f16 WMMA compute).  ACT: 0 = identity, 1 = exact GELU.
// Block = 256 threads (8 waves), tile 128x128, K-step 32.
// Staging: GLOBAL_LOAD_ASYNC_TO_LDS_B128 of f32 tiles, then f16 convert
// (B transposed to N-major for contiguous b128 fragment reads).
// ---------------------------------------------------------------------------
template <int ACT>
__global__ __launch_bounds__(256) void gemm_kernel(
    const float* __restrict__ A, const float* __restrict__ W,
    const float* __restrict__ bias, float* __restrict__ C,
    int M, int N, int K) {
  __shared__ __align__(16) float    Af32[128 * 32];  // async landing, 16KB
  __shared__ __align__(16) float    Bf32[32 * 128];  // async landing, 16KB
  __shared__ __align__(16) _Float16 As[128 * 32];    // row-major [m][k]
  __shared__ __align__(16) _Float16 Bs[128 * 32];    // N-major   [n][k]

  const int tid  = threadIdx.x;
  const int lane = tid & 31;
  const int wave = tid >> 5;
  const int bm   = blockIdx.y * 128;
  const int bn   = blockIdx.x * 128;
  const int row0 = wave * 16;

  v8f acc[8] = {};

  for (int k0 = 0; k0 < K; k0 += 32) {
    // ---- async stage A (128x32) and W (32x128) f32 tiles into LDS --------
#pragma unroll
    for (int t = 0; t < 4; ++t) {
      int c  = tid + t * 256;
      int ra = c >> 3, ca = (c & 7) * 4;   // A: 8 x b128 chunks per 32-wide row
      async_copy_b128(&Af32[ra * 32 + ca],
                      &A[(size_t)(bm + ra) * K + k0 + ca]);
      int rb = c >> 5, cb = (c & 31) * 4;  // W: 32 x b128 chunks per 128-wide row
      async_copy_b128(&Bf32[rb * 128 + cb],
                      &W[(size_t)(k0 + rb) * N + bn + cb]);
    }
    async_wait_all();
    __syncthreads();

    // ---- convert to f16 compute tiles ------------------------------------
    for (int i = tid; i < 128 * 32; i += 256) As[i] = (_Float16)Af32[i];
    for (int i = tid; i < 32 * 128; i += 256) {
      int r = i >> 7, c2 = i & 127;
      Bs[c2 * 32 + r] = (_Float16)Bf32[i];
    }
    __syncthreads();

    // ---- 8 waves x (1 A frag  x 8 B frags) = 64 WMMAs per K-step ---------
    v16h af = load_frag_a(As, 32, row0, 0, lane);
#pragma unroll
    for (int t = 0; t < 8; ++t) {
      v16h bf = load_frag_b(Bs, 32, t * 16, 0, lane);
      acc[t] = wmma_f16(af, bf, acc[t]);
    }
    __syncthreads();
  }

  // ---- epilogue: bias (+ GELU) and store ---------------------------------
#pragma unroll
  for (int t = 0; t < 8; ++t) {
    int col = bn + t * 16 + (lane & 15);
    float bv = bias[col];
    int rb = bm + row0 + ((lane & 16) ? 8 : 0);
#pragma unroll
    for (int r = 0; r < 8; ++r) {
      float v = acc[t][r] + bv;
      if (ACT == 1) v = 0.5f * v * (1.0f + erff(v * 0.70710678118654752f));
      C[(size_t)(rb + r) * N + col] = v;
    }
  }
}

// ---------------------------------------------------------------------------
// Attention kernel: one block (128 thr / 4 waves) per (b,n,head) group.
// Entire 64x64 q/k/v group resident in LDS; the three 64x64x64 products run
// on WMMA, the rank-8 dyn-proj / softmax pieces on VALU.
// ---------------------------------------------------------------------------
__global__ __launch_bounds__(128) void attn_kernel(
    const float* __restrict__ qkv, const float* __restrict__ ema,
    const float* __restrict__ dpk_w, const float* __restrict__ dpk_b,
    const float* __restrict__ dpv_w, const float* __restrict__ dpv_b,
    float* __restrict__ out_tok, float* __restrict__ out_hid) {
  __shared__ __align__(16) unsigned char smem[65536];
  _Float16* qT  = (_Float16*)(smem);            // [d][a]   8KB
  _Float16* ks  = (_Float16*)(smem + 8192);     // [a][d]   8KB
  _Float16* kT  = (_Float16*)(smem + 16384);    // [d][a]   8KB
  _Float16* vs  = (_Float16*)(smem + 24576);    // [a][d]   8KB
  _Float16* es  = (_Float16*)(smem + 32768);    // EMA f16; later attn_hid f16
  float* f32b = (float*)(smem + 40960);         // Eq, later scores_hid (16KB)
  float* pbuf = (float*)(smem + 57344);         // 64x8 logits / softmax
  float* kdp  = (float*)(smem + 59392);         // 8x64
  float* vdp  = (float*)(smem + 61440);         // 8x64
  float* ekdp = (float*)(smem + 63488);         // 8x64

  const int g    = blockIdx.x;
  const int bn   = g >> 3;
  const int head = g & 7;
  const int tid  = threadIdx.x;
  const int lane = tid & 31;
  const int wave = tid >> 5;
  const int row0 = wave * 16;
  const size_t t0 = (size_t)bn * 64;

  // ---- load q,k,v (64x64 each) + EMA into LDS (f16) -----------------------
  for (int i = tid; i < 4096; i += 128) {
    int a = i >> 6, d = i & 63;
    size_t base = (t0 + a) * (3 * DMODEL_) + head * HEADDIM_ + d;
    float qv = qkv[base];
    float kv = qkv[base + DMODEL_];
    float vv = qkv[base + 2 * DMODEL_];
    qT[d * 64 + a] = (_Float16)qv;
    ks[a * 64 + d] = (_Float16)kv;
    kT[d * 64 + a] = (_Float16)kv;
    vs[a * 64 + d] = (_Float16)vv;
    es[i] = (_Float16)ema[i];  // row-major 64x64
  }
  __syncthreads();

  // ---- dyn_proj for K: p = softmax(clip(k @ dpk_w + b)), kdp = p^T @ k ----
  for (int i = tid; i < 512; i += 128) {
    int e = i >> 3, c = i & 7;
    float s = dpk_b[c];
    for (int d = 0; d < 64; ++d) s += (float)ks[e * 64 + d] * dpk_w[d * 8 + c];
    pbuf[i] = fminf(30.0f, fmaxf(-30.0f, s));
  }
  __syncthreads();
  if (tid < 64) {
    float m = -1e30f;
    for (int c = 0; c < 8; ++c) m = fmaxf(m, pbuf[tid * 8 + c]);
    float s = 0.f;
    for (int c = 0; c < 8; ++c) { float e = __expf(pbuf[tid * 8 + c] - m); pbuf[tid * 8 + c] = e; s += e; }
    float inv = 1.0f / s;
    for (int c = 0; c < 8; ++c) pbuf[tid * 8 + c] *= inv;
  }
  __syncthreads();
  for (int i = tid; i < 512; i += 128) {
    int c = i >> 6, f = i & 63;
    float s = 0.f;
    for (int e = 0; e < 64; ++e) s += (float)ks[e * 64 + f] * pbuf[e * 8 + c];
    kdp[c * 64 + f] = s;
  }
  __syncthreads();

  // ---- dyn_proj for V -----------------------------------------------------
  for (int i = tid; i < 512; i += 128) {
    int e = i >> 3, c = i & 7;
    float s = dpv_b[c];
    for (int d = 0; d < 64; ++d) s += (float)vs[e * 64 + d] * dpv_w[d * 8 + c];
    pbuf[i] = fminf(30.0f, fmaxf(-30.0f, s));
  }
  __syncthreads();
  if (tid < 64) {
    float m = -1e30f;
    for (int c = 0; c < 8; ++c) m = fmaxf(m, pbuf[tid * 8 + c]);
    float s = 0.f;
    for (int c = 0; c < 8; ++c) { float e = __expf(pbuf[tid * 8 + c] - m); pbuf[tid * 8 + c] = e; s += e; }
    float inv = 1.0f / s;
    for (int c = 0; c < 8; ++c) pbuf[tid * 8 + c] *= inv;
  }
  __syncthreads();
  for (int i = tid; i < 512; i += 128) {
    int c = i >> 6, f = i & 63;
    float s = 0.f;
    for (int e = 0; e < 64; ++e) s += (float)vs[e * 64 + f] * pbuf[e * 8 + c];
    vdp[c * 64 + f] = s;
  }
  __syncthreads();

  // ---- Ekdp = ema[:8,:8] @ kdp -------------------------------------------
  for (int i = tid; i < 512; i += 128) {
    int c = i >> 6, f = i & 63;
    float s = 0.f;
    for (int j = 0; j < 8; ++j) s += ema[c * 64 + j] * kdp[j * 64 + f];
    ekdp[c * 64 + f] = s;
  }

  // ---- Eq = EMA(64x64) @ q(64x64)  [WMMA: A=es row-major, B=qT N-major] ---
#pragma unroll
  for (int t = 0; t < 4; ++t) {
    v8f acc = {};
#pragma unroll
    for (int kk = 0; kk < 2; ++kk) {
      v16h a = load_frag_a(es, 64, row0, kk * 32, lane);
      v16h b = load_frag_b(qT, 64, t * 16, kk * 32, lane);
      acc = wmma_f16(a, b, acc);
    }
    store_c_lds(f32b, 64, row0, t * 16, lane, acc, 1.0f);
  }
  __syncthreads();

  // ---- scores_tok = Eq @ Ekdp^T / 8, softmax(8), out_tok = attn @ vdp -----
  for (int i = tid; i < 512; i += 128) {
    int e = i >> 3, c = i & 7;
    float s = 0.f;
    for (int d = 0; d < 64; ++d) s += f32b[e * 64 + d] * ekdp[c * 64 + d];
    pbuf[i] = s * 0.125f;  // 1/sqrt(HEAD_DIM)
  }
  __syncthreads();
  if (tid < 64) {
    float m = -1e30f;
    for (int c = 0; c < 8; ++c) m = fmaxf(m, pbuf[tid * 8 + c]);
    float s = 0.f;
    for (int c = 0; c < 8; ++c) { float e = __expf(pbuf[tid * 8 + c] - m); pbuf[tid * 8 + c] = e; s += e; }
    float inv = 1.0f / s;
    for (int c = 0; c < 8; ++c) pbuf[tid * 8 + c] *= inv;
  }
  __syncthreads();
  for (int i = tid; i < 4096; i += 128) {
    int e = i >> 6, d = i & 63;
    float s = 0.f;
    for (int c = 0; c < 8; ++c) s += pbuf[e * 8 + c] * vdp[c * 64 + d];
    out_tok[(size_t)g * 4096 + i] = s;
  }
  __syncthreads();  // f32b (Eq) dead from here -> reuse for scores_hid

  // ---- scores_hid = q^T @ k / 8  [WMMA: A=qT row-major, B=kT N-major] -----
#pragma unroll
  for (int t = 0; t < 4; ++t) {
    v8f acc = {};
#pragma unroll
    for (int kk = 0; kk < 2; ++kk) {
      v16h a = load_frag_a(qT, 64, row0, kk * 32, lane);
      v16h b = load_frag_b(kT, 64, t * 16, kk * 32, lane);
      acc = wmma_f16(a, b, acc);
    }
    store_c_lds(f32b, 64, row0, t * 16, lane, acc, 0.125f);  // 1/sqrt(64)
  }
  __syncthreads();

  // row softmax over 64, write f16 attn into es (EMA copy now dead)
  _Float16* ah = es;
  if (tid < 64) {
    float m = -1e30f;
    for (int f = 0; f < 64; ++f) m = fmaxf(m, f32b[tid * 64 + f]);
    float s = 0.f;
    for (int f = 0; f < 64; ++f) { float e = __expf(f32b[tid * 64 + f] - m); f32b[tid * 64 + f] = e; s += e; }
    float inv = 1.0f / s;
    for (int f = 0; f < 64; ++f) ah[tid * 64 + f] = (_Float16)(f32b[tid * 64 + f] * inv);
  }
  __syncthreads();

  // ---- out_hid[a,e] = sum_f v[a,f] attn[e,f]  [A=vs, B=ah N-major] --------
#pragma unroll
  for (int t = 0; t < 4; ++t) {
    v8f acc = {};
#pragma unroll
    for (int kk = 0; kk < 2; ++kk) {
      v16h a = load_frag_a(vs, 64, row0, kk * 32, lane);
      v16h b = load_frag_b(ah, 64, t * 16, kk * 32, lane);
      acc = wmma_f16(a, b, acc);
    }
    int col = t * 16 + (lane & 15);
    int rb  = row0 + ((lane & 16) ? 8 : 0);
#pragma unroll
    for (int r = 0; r < 8; ++r)
      out_hid[(size_t)g * 4096 + (rb + r) * 64 + col] = acc[r];
  }
}

// ---------------------------------------------------------------------------
// merge(x) + LayerNorm.  One block per (bn, token) output row of 512.
// feat f -> c=f>>8, a=(f>>6)&3, d=f&63; head=2a+(tb>>5); e=2*(tb&31)+c.
// ---------------------------------------------------------------------------
__device__ __forceinline__ float block_sum(float v, float* red, int tid) {
  red[tid] = v;
  __syncthreads();
  for (int s = 64; s > 0; s >>= 1) {
    if (tid < s) red[tid] += red[tid + s];
    __syncthreads();
  }
  float r = red[0];
  __syncthreads();
  return r;
}

__global__ __launch_bounds__(128) void merge_ln_kernel(
    const float* __restrict__ x, const float* __restrict__ gam,
    const float* __restrict__ bet, float* __restrict__ out) {
  const int tb = blockIdx.x, bn = blockIdx.y, tid = threadIdx.x;
  __shared__ float red[128];
  float vals[4];
  float s = 0.f;
#pragma unroll
  for (int j = 0; j < 4; ++j) {
    int f = tid + j * 128;
    int c = f >> 8, a = (f >> 6) & 3, d = f & 63;
    int head = a * 2 + (tb >> 5);
    int e = ((tb & 31) << 1) | c;
    float v = x[(size_t)(bn * 8 + head) * 4096 + e * 64 + d];
    vals[j] = v;
    s += v;
  }
  float mean = block_sum(s, red, tid) * (1.0f / 512.0f);
  float q = 0.f;
#pragma unroll
  for (int j = 0; j < 4; ++j) { float d = vals[j] - mean; q += d * d; }
  float var  = block_sum(q, red, tid) * (1.0f / 512.0f);
  float rstd = rsqrtf(var + 1e-5f);
#pragma unroll
  for (int j = 0; j < 4; ++j) {
    int f = tid + j * 128;
    out[(size_t)(bn * 64 + tb) * 512 + f] = (vals[j] - mean) * rstd * gam[f] + bet[f];
  }
}

// ---------------------------------------------------------------------------
// out = LN(src + ffn1 + ffn2).  One block per token row.
// ---------------------------------------------------------------------------
__global__ __launch_bounds__(128) void final_ln_kernel(
    const float* __restrict__ src, const float* __restrict__ s1,
    const float* __restrict__ s2, const float* __restrict__ gam,
    const float* __restrict__ bet, float* __restrict__ out) {
  const int t = blockIdx.x, tid = threadIdx.x;
  __shared__ float red[128];
  float vals[4];
  float s = 0.f;
#pragma unroll
  for (int j = 0; j < 4; ++j) {
    size_t idx = (size_t)t * 512 + tid + j * 128;
    float v = src[idx] + s1[idx] + s2[idx];
    vals[j] = v;
    s += v;
  }
  float mean = block_sum(s, red, tid) * (1.0f / 512.0f);
  float q = 0.f;
#pragma unroll
  for (int j = 0; j < 4; ++j) { float d = vals[j] - mean; q += d * d; }
  float var  = block_sum(q, red, tid) * (1.0f / 512.0f);
  float rstd = rsqrtf(var + 1e-5f);
#pragma unroll
  for (int j = 0; j < 4; ++j) {
    int f = tid + j * 128;
    out[(size_t)t * 512 + f] = (vals[j] - mean) * rstd * gam[f] + bet[f];
  }
}

// ---------------------------------------------------------------------------
extern "C" void kernel_launch(void* const* d_in, const int* in_sizes, int n_in,
                              void* d_out, int out_size, void* d_ws,
                              size_t ws_size, hipStream_t stream) {
  const float* src    = (const float*)d_in[0];
  const float* ema    = (const float*)d_in[1];
  const float* qkv_w  = (const float*)d_in[2];
  const float* qkv_b  = (const float*)d_in[3];
  const float* dpk_w  = (const float*)d_in[4];
  const float* dpk_b  = (const float*)d_in[5];
  const float* dpv_w  = (const float*)d_in[6];
  const float* dpv_b  = (const float*)d_in[7];
  const float* ff1_w1 = (const float*)d_in[8];
  const float* ff1_b1 = (const float*)d_in[9];
  const float* ff1_w2 = (const float*)d_in[10];
  const float* ff1_b2 = (const float*)d_in[11];
  const float* ff2_w1 = (const float*)d_in[12];
  const float* ff2_b1 = (const float*)d_in[13];
  const float* ff2_w2 = (const float*)d_in[14];
  const float* ff2_b2 = (const float*)d_in[15];
  const float* np1_g  = (const float*)d_in[16];
  const float* np1_b  = (const float*)d_in[17];
  const float* np2_g  = (const float*)d_in[18];
  const float* np2_b  = (const float*)d_in[19];
  const float* na_g   = (const float*)d_in[20];
  const float* na_b   = (const float*)d_in[21];
  float* out = (float*)d_out;

  // workspace carve-up (floats)
  float* ws   = (float*)d_ws;
  float* qkv  = ws;                               // 21504*1536
  float* otok = qkv  + (size_t)MTOT_ * 1536;      // 2688*4096
  float* ohid = otok + (size_t)NGROUP_ * 4096;    // 2688*4096
  float* o1   = ohid + (size_t)NGROUP_ * 4096;    // 21504*512
  float* o2   = o1   + (size_t)MTOT_ * 512;       // 21504*512
  float* hbuf = o2   + (size_t)MTOT_ * 512;       // 21504*2048 (reused)
  float* sum1 = hbuf + (size_t)MTOT_ * 2048;      // 21504*512
  float* sum2 = sum1 + (size_t)MTOT_ * 512;       // 21504*512

  const dim3 gblk(256);
  const dim3 blk(128);

  // 1) QKV projection
  gemm_kernel<0><<<dim3(1536 / 128, MTOT_ / 128), gblk, 0, stream>>>(
      src, qkv_w, qkv_b, qkv, MTOT_, 1536, 512);

  // 2) dual attention per (b,n,head)
  attn_kernel<<<dim3(NGROUP_), blk, 0, stream>>>(
      qkv, ema, dpk_w, dpk_b, dpv_w, dpv_b, otok, ohid);

  // 3) merge + LN for both branches
  merge_ln_kernel<<<dim3(64, B_ * NVARS_), blk, 0, stream>>>(otok, np1_g, np1_b, o1);
  merge_ln_kernel<<<dim3(64, B_ * NVARS_), blk, 0, stream>>>(ohid, np2_g, np2_b, o2);

  // 4) FFN 1
  gemm_kernel<1><<<dim3(DFF_ / 128, MTOT_ / 128), gblk, 0, stream>>>(
      o1, ff1_w1, ff1_b1, hbuf, MTOT_, DFF_, 512);
  gemm_kernel<0><<<dim3(512 / 128, MTOT_ / 128), gblk, 0, stream>>>(
      hbuf, ff1_w2, ff1_b2, sum1, MTOT_, 512, DFF_);

  // 5) FFN 2
  gemm_kernel<1><<<dim3(DFF_ / 128, MTOT_ / 128), gblk, 0, stream>>>(
      o2, ff2_w1, ff2_b1, hbuf, MTOT_, DFF_, 512);
  gemm_kernel<0><<<dim3(512 / 128, MTOT_ / 128), gblk, 0, stream>>>(
      hbuf, ff2_w2, ff2_b2, sum2, MTOT_, 512, DFF_);

  // 6) residual + final LN
  final_ln_kernel<<<dim3(MTOT_), blk, 0, stream>>>(src, sum1, sum2, na_g, na_b, out);
}
